// CrossViTPointFusion_49194555408656
// MI455X (gfx1250) — matrix operand order repaired and checked by reading, not compiled, not run on hardware
//
#include <hip/hip_runtime.h>
#include <math.h>

// CrossViT point fusion for MI455X (gfx1250, wave32).
// Strategy: algebraically reduce k/v GEMMs to rank-8/rank-reduction passes
// (memory bound), and run the single unavoidable NxCxC GEMM (x_a @ Wo) with
// v_wmma_f32_16x16x32_bf16 so compute (~2 PFLOPS needed at 23.3 TB/s) keeps up.
// Round 2: stage packed Wo (128 KB) in LDS once per workgroup instead of
// re-streaming it from L2 per wave (8x less B traffic; WMMA feeds from DS).

typedef __attribute__((ext_vector_type(16))) __bf16 v16bf;
typedef __attribute__((ext_vector_type(8)))  float  v8f;

#define C   256
#define HH  8
#define DH  32

union V16 { v16bf v; __bf16 e[16]; };

__device__ inline int seg_of(int n, const long long* off, int B) {
  int s = 0;
  while (s < B - 1 && (long long)n >= off[s]) ++s;
  return s;
}

__device__ inline void atomicMaxF(float* a, float v) {
  if (v >= 0.f) atomicMax((int*)a, __float_as_int(v));
  else          atomicMin((unsigned int*)a, __float_as_uint(v));
}

// ---------------------------------------------------------------- init
__global__ __launch_bounds__(256) void k_init(float* cls_sum, float* segmax,
                                              float* segden, float* U) {
  int i = blockIdx.x * 256 + threadIdx.x;
  if (i < 16 * C)       cls_sum[i] = 0.f;
  if (i < 16 * HH)      { segmax[i] = -__builtin_inff(); segden[i] = 0.f; }
  if (i < 16 * HH * C)  U[i] = 0.f;
}

// ---------------------------------------------------------------- pack Wo -> bf16 WMMA B-operand layout
// WoB[ct][s][lane][e] : lane holds column n=lane%16, K = s*32 + (lane<16?0:8) + (e&7) + (e<8?0:16)
__global__ __launch_bounds__(256) void k_pack(const float* __restrict__ Wo,
                                              __bf16* __restrict__ WoB) {
  int idx = blockIdx.x * 256 + threadIdx.x;
  if (idx >= 16 * 8 * 32 * 16) return;
  int e    = idx & 15;
  int lane = (idx >> 4) & 31;
  int s    = (idx >> 9) & 7;
  int ct   = idx >> 12;
  int j    = s * 32 + ((lane & 16) ? 8 : 0) + (e & 7) + ((e & 8) ? 16 : 0);   // K index
  int c    = ct * 16 + (lane & 15);                 // N index
  WoB[idx] = (__bf16)Wo[j * C + c];
}

// ---------------------------------------------------------------- segment mean-pool partials of x_a
__global__ __launch_bounds__(256) void k_pool(const float* __restrict__ x_a,
                                              const long long* __restrict__ off,
                                              int B, int N, float* __restrict__ cls_sum) {
  int c = threadIdx.x;
  int row0 = blockIdx.x * 512;
  if (row0 >= N) return;
  int rend = row0 + 512; if (rend > N) rend = N;
  int seg = seg_of(row0, off, B);
  long long bnd = off[seg];
  float acc = 0.f;
  for (int n = row0; n < rend; ++n) {
    if ((long long)n >= bnd) {
      atomicAdd(&cls_sum[seg * C + c], acc);
      acc = 0.f;
      do { ++seg; } while (seg < B - 1 && off[seg] <= (long long)n);
      bnd = off[seg];
    }
    acc += x_a[(size_t)n * C + c];
  }
  atomicAdd(&cls_sum[seg * C + c], acc);
}

// ---------------------------------------------------------------- cls -> q -> KA (tiny)
__global__ __launch_bounds__(256) void k_small1(const float* __restrict__ Wq,
                                                const float* __restrict__ Wk,
                                                const long long* __restrict__ off, int B,
                                                const float* __restrict__ cls_sum,
                                                float* __restrict__ KA) {
  __shared__ float cls[16 * C];
  __shared__ float qs[16 * C];
  int tid = threadIdx.x;
  for (int idx = tid; idx < 16 * C; idx += 256) {
    int b = idx >> 8;
    float v = 0.f;
    if (b < B) {
      float cnt = (float)(off[b] - (b ? off[b - 1] : 0));
      v = cls_sum[idx] / cnt;
    }
    cls[idx] = v;
  }
  __syncthreads();
  for (int idx = tid; idx < 16 * C; idx += 256) {
    int b = idx >> 8, c = idx & 255;
    float s = 0.f;
    for (int j = 0; j < C; ++j) s += cls[(b << 8) + j] * Wq[j * C + c];
    qs[idx] = s;
  }
  __syncthreads();
  const float scale = 0.17677669529663687f;  // 1/sqrt(32)
  for (int idx = tid; idx < 16 * HH * C; idx += 256) {
    int b = idx >> 11;
    int h = (idx >> 8) & 7;
    int j = idx & 255;
    float s = 0.f;
#pragma unroll
    for (int d = 0; d < DH; ++d)
      s += Wk[j * C + h * DH + d] * qs[(b << 8) + h * DH + d];
    KA[idx] = s * scale;  // KA[b][h][j]
  }
}

// ---------------------------------------------------------------- logits = x_b . KA[seg] ; segment max
__global__ __launch_bounds__(256) void k_logits(const float* __restrict__ x_b,
                                                const float* __restrict__ KA,
                                                const long long* __restrict__ off,
                                                int B, int N,
                                                float* __restrict__ logits,
                                                float* __restrict__ segmax) {
  __shared__ float xs[32][C + 4];  // padded: no LDS bank conflicts
  int tid = threadIdx.x;
  int row0 = blockIdx.x * 32;
  for (int i = tid; i < 32 * C; i += 256) {
    int r = i >> 8, c = i & 255;
    int n = row0 + r;
    xs[r][c] = (n < N) ? x_b[(size_t)n * C + c] : 0.f;
  }
  __syncthreads();
  int r = tid >> 3, h = tid & 7;
  int n = row0 + r;
  if (n >= N) return;
  int seg = seg_of(n, off, B);
  const float* ka = KA + (size_t)(seg * HH + h) * C;
  float s = 0.f;
#pragma unroll 8
  for (int j = 0; j < C; ++j) s += xs[r][j] * ka[j];
  logits[(size_t)n * HH + h] = s;
  atomicMaxF(&segmax[seg * HH + h], s);
}

// ---------------------------------------------------------------- p = exp(l - max); segment denom
__global__ __launch_bounds__(256) void k_pexp(const long long* __restrict__ off, int B, int N,
                                              const float* __restrict__ segmax,
                                              float* __restrict__ logits,
                                              float* __restrict__ segden) {
  __shared__ float part[16 * HH];
  int tid = threadIdx.x;
  for (int i = tid; i < 16 * HH; i += 256) part[i] = 0.f;
  __syncthreads();
  int row0 = blockIdx.x * 256;
  for (int idx = tid; idx < 256 * HH; idx += 256) {
    int n = row0 + (idx >> 3);
    int h = idx & 7;
    if (n >= N) continue;
    int seg = seg_of(n, off, B);
    float p = __expf(logits[(size_t)n * HH + h] - segmax[seg * HH + h]);
    logits[(size_t)n * HH + h] = p;  // store p in place
    atomicAdd(&part[seg * HH + h], p);
  }
  __syncthreads();
  for (int i = tid; i < 16 * HH; i += 256)
    if (part[i] != 0.f) atomicAdd(&segden[i], part[i]);
}

// ---------------------------------------------------------------- U[b,h,j] = sum_n p[n,h] * x_b[n,j]
__global__ __launch_bounds__(256) void k_uacc(const float* __restrict__ x_b,
                                              const long long* __restrict__ off,
                                              int B, int N,
                                              const float* __restrict__ p,
                                              float* __restrict__ U) {
  __shared__ float pt[512 * HH];
  int tid = threadIdx.x;
  int row0 = blockIdx.x * 512;
  if (row0 >= N) return;
  int rend = row0 + 512; if (rend > N) rend = N;
  int nr = rend - row0;
  for (int i = tid; i < nr * HH; i += 256) pt[i] = p[(size_t)row0 * HH + i];
  __syncthreads();
  int c = tid;
  int seg = seg_of(row0, off, B);
  long long bnd = off[seg];
  float acc[HH];
#pragma unroll
  for (int h = 0; h < HH; ++h) acc[h] = 0.f;
  for (int r = 0; r < nr; ++r) {
    int n = row0 + r;
    if ((long long)n >= bnd) {
#pragma unroll
      for (int h = 0; h < HH; ++h) {
        atomicAdd(&U[(size_t)(seg * HH + h) * C + c], acc[h]);
        acc[h] = 0.f;
      }
      do { ++seg; } while (seg < B - 1 && off[seg] <= (long long)n);
      bnd = off[seg];
    }
    float xv = x_b[(size_t)n * C + c];  // coalesced: thread==column
#pragma unroll
    for (int h = 0; h < HH; ++h) acc[h] += pt[r * HH + h] * xv;
  }
#pragma unroll
  for (int h = 0; h < HH; ++h) atomicAdd(&U[(size_t)(seg * HH + h) * C + c], acc[h]);
}

// ---------------------------------------------------------------- out_cls -> proj -> MLP -> D (tiny)
__global__ __launch_bounds__(256) void k_small2(const float* Wv, const float* Wp, const float* bp,
                                                const float* Wr1, const float* br1,
                                                const float* Wr2, const float* br2,
                                                const float* gamma, const float* Wo, const float* bo,
                                                const float* __restrict__ U,
                                                const float* __restrict__ segden, int B,
                                                float* __restrict__ Dbuf) {
  __shared__ float s0[16 * C];
  __shared__ float s1[16 * C];
  int tid = threadIdx.x;
  for (int idx = tid; idx < 16 * C; idx += 256) {  // out_cls = (U @ Wv)/den
    int b = idx >> 8, c = idx & 255, h = c >> 5;
    float s = 0.f;
    for (int j = 0; j < C; ++j) s += U[(size_t)((b << 3) + h) * C + j] * Wv[j * C + c];
    s0[idx] = (b < B) ? s / segden[(b << 3) + h] : 0.f;
  }
  __syncthreads();
  for (int idx = tid; idx < 16 * C; idx += 256) {  // @Wp + bp
    int b = idx >> 8, c = idx & 255;
    float s = bp[c];
    for (int j = 0; j < C; ++j) s += s0[(b << 8) + j] * Wp[j * C + c];
    s1[idx] = s;
  }
  __syncthreads();
  for (int idx = tid; idx < 16 * C; idx += 256) {  // relu(@Wr1 + br1)
    int b = idx >> 8, c = idx & 255;
    float s = br1[c];
    for (int j = 0; j < C; ++j) s += s1[(b << 8) + j] * Wr1[j * C + c];
    s0[idx] = fmaxf(s, 0.f);
  }
  __syncthreads();
  for (int idx = tid; idx < 16 * C; idx += 256) {  // @Wr2 + br2 = cls_proj
    int b = idx >> 8, c = idx & 255;
    float s = br2[c];
    for (int j = 0; j < C; ++j) s += s0[(b << 8) + j] * Wr2[j * C + c];
    s1[idx] = s;
  }
  __syncthreads();
  float g = gamma[0];
  for (int idx = tid; idx < 16 * C; idx += 256) {  // D = g*(cls_proj@Wo) + bo
    int b = idx >> 8, c = idx & 255;
    float s = 0.f;
    for (int j = 0; j < C; ++j) s += s1[(b << 8) + j] * Wo[j * C + c];
    Dbuf[idx] = g * s + bo[c];
  }
}

// ---------------------------------------------------------------- final GEMM: out = x_a @ Wo + D[seg]
// ISA 16-bit A layout (16x32 slice): lanes 0-15 hold M=lane, K={base..base+7, base+16..23};
// lanes 16-31 hold M=lane-16, K={base+8..15, base+24..31}.
__device__ inline v16bf load_a(const float* rowptr, int s, int lane) {
  int koff = s * 32 + ((lane & 16) ? 8 : 0);
  float4 f0 = *(const float4*)(rowptr + koff);
  float4 f1 = *(const float4*)(rowptr + koff + 4);
  float4 f2 = *(const float4*)(rowptr + koff + 16);
  float4 f3 = *(const float4*)(rowptr + koff + 20);
  V16 r;
  r.e[0]  = (__bf16)f0.x; r.e[1]  = (__bf16)f0.y; r.e[2]  = (__bf16)f0.z; r.e[3]  = (__bf16)f0.w;
  r.e[4]  = (__bf16)f1.x; r.e[5]  = (__bf16)f1.y; r.e[6]  = (__bf16)f1.z; r.e[7]  = (__bf16)f1.w;
  r.e[8]  = (__bf16)f2.x; r.e[9]  = (__bf16)f2.y; r.e[10] = (__bf16)f2.z; r.e[11] = (__bf16)f2.w;
  r.e[12] = (__bf16)f3.x; r.e[13] = (__bf16)f3.y; r.e[14] = (__bf16)f3.z; r.e[15] = (__bf16)f3.w;
  return r.v;
}

__global__ __launch_bounds__(256) void k_gemm(const float* __restrict__ x_a,
                                              const __bf16* __restrict__ WoB,
                                              const float* __restrict__ Dbuf,
                                              const long long* __restrict__ off,
                                              int B, int N, float* __restrict__ out) {
  // 144 KB LDS per workgroup (gfx1250 WGP: 320 KB) -> 2 workgroups/WGP.
  __shared__ __bf16 Bs[16 * 8 * 32 * 16];  // 128 KB: full packed Wo, staged once
  __shared__ float  Dl[16 * C];            // 16 KB: per-segment epilogue rows
  int tid = threadIdx.x;
  for (int i = tid; i < 16 * C; i += 256) Dl[i] = Dbuf[i];
  {
    const float4* src = (const float4*)WoB;
    float4*       dst = (float4*)Bs;
    for (int i = tid; i < (16 * 8 * 32 * 16) / 8; i += 256) dst[i] = src[i];
  }
  __syncthreads();

  int wave = tid >> 5, lane = tid & 31;
  int rowbase0 = blockIdx.x * 256 + wave * 32;  // each wave: 2 row-tiles (B-operand reuse)
  if (rowbase0 >= N) return;

  const float* rp0 = x_a + (size_t)(rowbase0 + (lane & 15)) * C;
  const float* rp1 = rp0 + (size_t)16 * C;
  v16bf a0[8], a1[8];
#pragma unroll
  for (int s = 0; s < 8; ++s) { a0[s] = load_a(rp0, s, lane); a1[s] = load_a(rp1, s, lane); }

  int m0 = rowbase0 + ((lane & 16) ? 8 : 0);  // first output row this lane owns
  int dof0[8], dof1[8];
#pragma unroll
  for (int r = 0; r < 8; ++r) {
    dof0[r] = seg_of(m0 + r, off, B) * C;
    dof1[r] = seg_of(m0 + 16 + r, off, B) * C;
  }

  const v16bf* Bv = (const v16bf*)Bs;
  for (int ct = 0; ct < 16; ++ct) {
    v8f c0 = {0.f, 0.f, 0.f, 0.f, 0.f, 0.f, 0.f, 0.f};
    v8f c1 = {0.f, 0.f, 0.f, 0.f, 0.f, 0.f, 0.f, 0.f};
#pragma unroll
    for (int s = 0; s < 8; ++s) {
      v16bf bm = Bv[(ct * 8 + s) * 32 + lane];  // 32 B/lane from LDS
      c0 = __builtin_amdgcn_wmma_f32_16x16x32_bf16(false, a0[s], false, bm, (short)0, c0, false, false);
      c1 = __builtin_amdgcn_wmma_f32_16x16x32_bf16(false, a1[s], false, bm, (short)0, c1, false, false);
    }
    int col = ct * 16 + (lane & 15);
#pragma unroll
    for (int r = 0; r < 8; ++r) {
      out[(size_t)(m0 + r) * C + col]      = c0[r] + Dl[dof0[r] + col];
      out[(size_t)(m0 + 16 + r) * C + col] = c1[r] + Dl[dof1[r] + col];
    }
  }
}

// ---------------------------------------------------------------- host
extern "C" void kernel_launch(void* const* d_in, const int* in_sizes, int n_in,
                              void* d_out, int out_size, void* d_ws, size_t ws_size,
                              hipStream_t stream) {
  const float*     x_a   = (const float*)d_in[0];
  const float*     x_b   = (const float*)d_in[1];
  const long long* off   = (const long long*)d_in[2];
  const float*     Wq    = (const float*)d_in[3];
  const float*     Wk    = (const float*)d_in[4];
  const float*     Wv    = (const float*)d_in[5];
  const float*     Wp    = (const float*)d_in[6];
  const float*     bp    = (const float*)d_in[7];
  const float*     Wr1   = (const float*)d_in[8];
  const float*     br1   = (const float*)d_in[9];
  const float*     Wr2   = (const float*)d_in[10];
  const float*     br2   = (const float*)d_in[11];
  const float*     gamma = (const float*)d_in[12];
  const float*     Wo    = (const float*)d_in[13];
  const float*     bo    = (const float*)d_in[14];
  float* out = (float*)d_out;

  int N = in_sizes[0] / C;
  int B = in_sizes[2];

  // workspace layout (floats): cls_sum 4K | KA 32K | segmax 128 | segden 128 | U 32K | D 4K | WoB(bf16) 64K | logits N*8
  float* ws       = (float*)d_ws;
  float* cls_sum  = ws;
  float* KA       = ws + 16 * C;
  float* segmax   = KA + 16 * HH * C;
  float* segden   = segmax + 16 * HH;
  float* U        = segden + 16 * HH;
  float* Dbuf     = U + 16 * HH * C;
  __bf16* WoB     = (__bf16*)(Dbuf + 16 * C);
  float* logits   = (float*)(WoB + 16 * 8 * 32 * 16);

  k_init  <<<128, 256, 0, stream>>>(cls_sum, segmax, segden, U);
  k_pack  <<<256, 256, 0, stream>>>(Wo, WoB);
  k_pool  <<<(N + 511) / 512, 256, 0, stream>>>(x_a, off, B, N, cls_sum);
  k_small1<<<1, 256, 0, stream>>>(Wq, Wk, off, B, cls_sum, KA);
  k_logits<<<(N + 31) / 32, 256, 0, stream>>>(x_b, KA, off, B, N, logits, segmax);
  k_pexp  <<<(N + 255) / 256, 256, 0, stream>>>(off, B, N, segmax, logits, segden);
  k_uacc  <<<(N + 511) / 512, 256, 0, stream>>>(x_b, off, B, N, logits, U);
  k_small2<<<1, 256, 0, stream>>>(Wv, Wp, bp, Wr1, br1, Wr2, br2, gamma, Wo, bo, U, segden, B, Dbuf);
  k_gemm  <<<(N + 255) / 256, 256, 0, stream>>>(x_a, WoB, Dbuf, off, B, N, out);
}
